// Attention_20418274525252
// MI455X (gfx1250) — matrix operand order repaired and checked by reading, not compiled, not run on hardware
//
#include <hip/hip_runtime.h>
#include <hip/hip_bf16.h>

#define DIMM   1024
#define HEADS  16
#define DHEAD  64
#define INNER  1024
#define BSZ    2
#define SEQ    2048
#define NTOK   (BSZ*SEQ)          // 4096
#define QKSCALE 0.125f            // 64^-0.5

typedef _Float16 f16;
typedef __attribute__((ext_vector_type(16))) _Float16 v16h;
typedef __attribute__((ext_vector_type(8)))  _Float16 v8h;
typedef __attribute__((ext_vector_type(8)))  float    v8f;

union F16x16 { v16h v; v8h h[2]; };

// ---------------------------------------------------------------------------
// WMMA helper (wave32, gfx1250). D = A(16x32 f16) * B(32x16 f16) + C(16x16 f32)
// ---------------------------------------------------------------------------
__device__ inline v8f wmma_f16(v16h a, v16h b, v8f c) {
    return __builtin_amdgcn_wmma_f32_16x16x32_f16(
        /*neg_a=*/false, a, /*neg_b=*/false, b,
        /*c_mod=*/(short)0, c, /*reuse_a=*/false, /*reuse_b=*/false);
}

// A fragment (16x32 f16) from row-major tile, ld in halves.
// Lane l: row = l%16 ; K = kb..kb+7 (v0..3) and kb+16..kb+23 (v4..7), kb=(l/16)*8
__device__ inline v16h load_a_frag(const f16* base, int ld) {
    int lane = threadIdx.x & 31;
    int row  = lane & 15;
    int kb   = (lane >> 4) << 3;
    const f16* p = base + row * ld + kb;
    F16x16 r;
    r.h[0] = *(const v8h*)(p);
    r.h[1] = *(const v8h*)(p + 16);
    return r.v;
}

// B fragment (32x16 f16) from TRANSPOSED source Bt[n][k] (k contiguous).
// Lane l: col n = l%16 ; K = kb..kb+7 (v0..3) and kb+16..kb+23 (v4..7), kb=(l/16)*8
__device__ inline v16h load_bt_frag(const f16* bt, int ld) {
    int lane = threadIdx.x & 31;
    int col  = lane & 15;
    int kb   = (lane >> 4) << 3;
    const f16* p = bt + col * ld + kb;
    F16x16 r;
    r.h[0] = *(const v8h*)(p);
    r.h[1] = *(const v8h*)(p + 16);
    return r.v;
}

// Async global->LDS 16-byte copy (CDNA5 GLOBAL_LOAD_ASYNC_TO_LDS_B128, ASYNCcnt)
__device__ inline void async_copy_b128(unsigned lds_byte_off, const f16* gptr) {
    asm volatile("global_load_async_to_lds_b128 %0, %1, off"
                 :: "v"(lds_byte_off), "v"(gptr) : "memory");
}

// ---------------------------------------------------------------------------
// LayerNorm (f32 in) -> f16 normalized activations
// ---------------------------------------------------------------------------
__global__ __launch_bounds__(256) void ln_to_f16(const float* __restrict__ x,
                                                 const float* __restrict__ g,
                                                 const float* __restrict__ be,
                                                 f16* __restrict__ xn) {
    int row = blockIdx.x;                       // 0..NTOK-1
    const float* xr = x + (size_t)row * DIMM;
    float v[4], s = 0.f, ss = 0.f;
#pragma unroll
    for (int i = 0; i < 4; ++i) {
        v[i] = xr[threadIdx.x + 256 * i];
        s += v[i]; ss += v[i] * v[i];
    }
#pragma unroll
    for (int d = 16; d >= 1; d >>= 1) {
        s  += __shfl_xor(s,  d, 32);
        ss += __shfl_xor(ss, d, 32);
    }
    __shared__ float rs[8], rss[8];
    int wv = threadIdx.x >> 5;
    if ((threadIdx.x & 31) == 0) { rs[wv] = s; rss[wv] = ss; }
    __syncthreads();
    s = 0.f; ss = 0.f;
#pragma unroll
    for (int i = 0; i < 8; ++i) { s += rs[i]; ss += rss[i]; }
    float mu  = s * (1.f / DIMM);
    float var = ss * (1.f / DIMM) - mu * mu;
    float inv = rsqrtf(var + 1e-5f);
    f16* xo = xn + (size_t)row * DIMM;
#pragma unroll
    for (int i = 0; i < 4; ++i) {
        int c = threadIdx.x + 256 * i;
        xo[c] = (f16)((v[i] - mu) * inv * g[c] + be[c]);
    }
}

// ---------------------------------------------------------------------------
// Weight convert+transpose: W[k][n] f32 -> Wt[n][k] f16  (one-time prep)
// ---------------------------------------------------------------------------
__global__ void wt_transpose(const float* __restrict__ W, f16* __restrict__ Wt,
                             int K, int Ncol) {
    int idx = blockIdx.x * 256 + threadIdx.x;   // over output Wt
    if (idx < K * Ncol) {
        int n = idx / K, k = idx - n * K;
        Wt[idx] = (f16)W[(size_t)k * Ncol + n];
    }
}

// ---------------------------------------------------------------------------
// QKV projection: each wave computes a 16x64 strip (1 A frag -> 4 WMMAs).
// grid (NTOK/64, 3072/64), block 128.
// ---------------------------------------------------------------------------
__global__ __launch_bounds__(128) void qkv_gemm(const f16* __restrict__ xn,
                                                const f16* __restrict__ wt,
                                                f16* __restrict__ q,
                                                f16* __restrict__ k,
                                                f16* __restrict__ vt) {
    int wave = threadIdx.x >> 5, lane = threadIdx.x & 31;
    int mtile = blockIdx.x * 4 + wave;
    int n0 = blockIdx.y * 64;
    const f16* a_base = xn + (size_t)(mtile * 16) * DIMM;
    const f16* b_base = wt + (size_t)n0 * DIMM;
    v8f acc0 = {}, acc1 = {}, acc2 = {}, acc3 = {};
#pragma unroll 2
    for (int k0 = 0; k0 < DIMM; k0 += 32) {
        v16h a = load_a_frag(a_base + k0, DIMM);
        acc0 = wmma_f16(a, load_bt_frag(b_base + (size_t) 0 * DIMM + k0, DIMM), acc0);
        acc1 = wmma_f16(a, load_bt_frag(b_base + (size_t)16 * DIMM + k0, DIMM), acc1);
        acc2 = wmma_f16(a, load_bt_frag(b_base + (size_t)32 * DIMM + k0, DIMM), acc2);
        acc3 = wmma_f16(a, load_bt_frag(b_base + (size_t)48 * DIMM + k0, DIMM), acc3);
    }
    int m_local = (lane >> 4) * 8, n_local = lane & 15;
    v8f accs[4] = {acc0, acc1, acc2, acc3};
#pragma unroll
    for (int j = 0; j < 4; ++j) {
        int col = n0 + j * 16 + n_local;
        int region = col >> 10;                 // 0=Q 1=K 2=V
        int ci = col & 1023, h = ci >> 6, d = ci & 63;
#pragma unroll
        for (int r = 0; r < 8; ++r) {
            int tok = mtile * 16 + m_local + r;
            int b_  = tok >> 11, ntok = tok & (SEQ - 1);
            int bh  = b_ * HEADS + h;
            float val = accs[j][r];
            if (region == 0)
                q[(((size_t)bh * SEQ) + ntok) * DHEAD + d] = (f16)(val * QKSCALE);
            else if (region == 1)
                k[(((size_t)bh * SEQ) + ntok) * DHEAD + d] = (f16)val;
            else
                vt[(((size_t)bh * DHEAD) + d) * SEQ + ntok] = (f16)val;
        }
    }
}

// ---------------------------------------------------------------------------
// Flash attention. Block = 4 waves, same (b,h), 64 query rows. K/V 32-key
// blocks are double-buffered in LDS via GLOBAL_LOAD_ASYNC_TO_LDS_B128 and
// shared by all 4 waves. Online softmax with per-lane row stats; P tile
// re-laid-out through LDS (C-layout -> A-layout).
// grid (SEQ/64, B*HEADS), block 128.
// ---------------------------------------------------------------------------
#define NPAD 48       // P tile row stride (halves): 96B, 16B-aligned
#define KLD  72       // K LDS row stride (halves): 144B -> conflict-free frags
#define VLD  40       // V LDS row stride (halves): 80B  -> conflict-free frags

__global__ __launch_bounds__(128) void attn(const f16* __restrict__ q,
                                            const f16* __restrict__ k,
                                            const f16* __restrict__ vt,
                                            f16* __restrict__ o) {
    __shared__ __align__(16) f16 kbufS[2][32 * KLD];   // 9.0 KB
    __shared__ __align__(16) f16 vbufS[2][64 * VLD];   // 10.0 KB
    __shared__ __align__(16) f16 pbufS[4][16 * NPAD];  // 6.0 KB
    int tid  = threadIdx.x;
    int wave = tid >> 5, lane = tid & 31;
    int bh    = blockIdx.y;                     // b*HEADS+h
    int qtile = blockIdx.x * 4 + wave;
    const f16* qb  = q  + ((size_t)bh * SEQ + qtile * 16) * DHEAD;
    const f16* kbb = k  + (size_t)bh * SEQ * DHEAD;
    const f16* vbb = vt + (size_t)bh * DHEAD * SEQ;

    v16h qa0 = load_a_frag(qb,      DHEAD);     // d 0..31 (scale pre-folded)
    v16h qa1 = load_a_frag(qb + 32, DHEAD);     // d 32..63

    v8f o0 = {}, o1 = {}, o2 = {}, o3 = {};
    float mi[8], li[8];
#pragma unroll
    for (int r = 0; r < 8; ++r) { mi[r] = -3.0e30f; li[r] = 0.f; }

    unsigned kbase = (unsigned)(uintptr_t)(&kbufS[0][0]);
    unsigned vbase = (unsigned)(uintptr_t)(&vbufS[0][0]);

    // stage one 32-key K/V block into LDS buffer `buf` (4 async b128 / thread)
    auto stage = [&](int buf, int kb0) {
        // K: 32 rows x 128B, padded rows. 4 threads per row, 2 b128 each.
        int krow = tid >> 2, kq = tid & 3;
        const f16* gk = kbb + (size_t)(kb0 + krow) * DHEAD + kq * 16;
        unsigned lk = kbase + (unsigned)buf * (32 * KLD * 2) + krow * (KLD * 2) + kq * 32;
        async_copy_b128(lk,      gk);
        async_copy_b128(lk + 16, gk + 8);
        // V: 64 rows (d-chan) x 64B, padded rows. 2 threads per row, 2 b128 each.
        int dch = tid >> 1, half = tid & 1;
        const f16* gv = vbb + (size_t)dch * SEQ + kb0 + half * 16;
        unsigned lv = vbase + (unsigned)buf * (64 * VLD * 2) + dch * (VLD * 2) + half * 32;
        async_copy_b128(lv,      gv);
        async_copy_b128(lv + 16, gv + 8);
    };

    f16* pp = pbufS[wave];
    int arow = lane & 15, acb = (lane >> 4) << 3;   // A-frag read pattern
    int prow = (lane >> 4) * 8, pcol = lane & 15;   // C-layout write pattern

    stage(0, 0);
    for (int kb0 = 0; kb0 < SEQ; kb0 += 32) {
        int  cur  = (kb0 >> 5) & 1;
        bool more = (kb0 + 32) < SEQ;
        if (more) {
            stage(cur ^ 1, kb0 + 32);               // prefetch next block
            asm volatile("s_wait_asynccnt 4" ::: "memory");  // oldest 4 done
        } else {
            asm volatile("s_wait_asynccnt 0" ::: "memory");
        }
        __syncthreads();                            // cur buffer visible to all

        const f16* kl = &kbufS[cur][0];
        const f16* vl = &vbufS[cur][0];

        // S = Q K^T for 32 keys (two 16-key C tiles, K-dim = d = 64)
        v8f s0 = {}, s1 = {};
        s0 = wmma_f16(qa0, load_bt_frag(kl +  0 * KLD +  0, KLD), s0);
        s0 = wmma_f16(qa1, load_bt_frag(kl +  0 * KLD + 32, KLD), s0);
        s1 = wmma_f16(qa0, load_bt_frag(kl + 16 * KLD +  0, KLD), s1);
        s1 = wmma_f16(qa1, load_bt_frag(kl + 16 * KLD + 32, KLD), s1);

        // online softmax: row stats across 16-lane half-groups
#pragma unroll
        for (int r = 0; r < 8; ++r) {
            float mx = fmaxf(s0[r], s1[r]);
#pragma unroll
            for (int d = 8; d >= 1; d >>= 1) mx = fmaxf(mx, __shfl_xor(mx, d, 16));
            float mn   = fmaxf(mi[r], mx);
            float corr = __expf(mi[r] - mn);
            float p0 = __expf(s0[r] - mn), p1 = __expf(s1[r] - mn);
            float rsum = p0 + p1;
#pragma unroll
            for (int d = 8; d >= 1; d >>= 1) rsum += __shfl_xor(rsum, d, 16);
            li[r] = li[r] * corr + rsum;
            mi[r] = mn;
            o0[r] *= corr; o1[r] *= corr; o2[r] *= corr; o3[r] *= corr;
            pp[(prow + r) * NPAD + pcol]      = (f16)p0;
            pp[(prow + r) * NPAD + pcol + 16] = (f16)p1;
        }
        asm volatile("s_wait_dscnt 0" ::: "memory");  // cross-lane LDS RAW (same wave)

        // P tile as A fragment (16x32)
        F16x16 pa;
        pa.h[0] = *(const v8h*)(pp + arow * NPAD + acb);
        pa.h[1] = *(const v8h*)(pp + arow * NPAD + acb + 16);
        v16h pav = pa.v;

        // O += P V  (4 d-chunks of 16)
        o0 = wmma_f16(pav, load_bt_frag(vl +  0 * VLD, VLD), o0);
        o1 = wmma_f16(pav, load_bt_frag(vl + 16 * VLD, VLD), o1);
        o2 = wmma_f16(pav, load_bt_frag(vl + 32 * VLD, VLD), o2);
        o3 = wmma_f16(pav, load_bt_frag(vl + 48 * VLD, VLD), o3);

        __syncthreads();                            // all readers done before overwrite
    }

    // epilogue: normalize, store O as [b][n][h*64+d] f16 for the out-proj GEMM
    int b_ = bh >> 4, h = bh & 15;
#pragma unroll
    for (int r = 0; r < 8; ++r) {
        float inv = 1.0f / li[r];
        int tok = qtile * 16 + prow + r;
        f16* orow = o + ((size_t)(b_ * SEQ + tok)) * INNER + h * DHEAD;
        orow[pcol +  0] = (f16)(o0[r] * inv);
        orow[pcol + 16] = (f16)(o1[r] * inv);
        orow[pcol + 32] = (f16)(o2[r] * inv);
        orow[pcol + 48] = (f16)(o3[r] * inv);
    }
}

// ---------------------------------------------------------------------------
// Output projection: wave computes 16x64 strip; out = O @ W_out + b_out (f32)
// grid (NTOK/64, DIMM/64), block 128
// ---------------------------------------------------------------------------
__global__ __launch_bounds__(128) void out_gemm(const f16* __restrict__ oin,
                                                const f16* __restrict__ wt,
                                                const float* __restrict__ bias,
                                                float* __restrict__ out) {
    int wave = threadIdx.x >> 5, lane = threadIdx.x & 31;
    int mtile = blockIdx.x * 4 + wave;
    int n0 = blockIdx.y * 64;
    const f16* ab = oin + (size_t)(mtile * 16) * INNER;
    const f16* bb = wt + (size_t)n0 * INNER;
    v8f acc0 = {}, acc1 = {}, acc2 = {}, acc3 = {};
#pragma unroll 2
    for (int k0 = 0; k0 < INNER; k0 += 32) {
        v16h a = load_a_frag(ab + k0, INNER);
        acc0 = wmma_f16(a, load_bt_frag(bb + (size_t) 0 * INNER + k0, INNER), acc0);
        acc1 = wmma_f16(a, load_bt_frag(bb + (size_t)16 * INNER + k0, INNER), acc1);
        acc2 = wmma_f16(a, load_bt_frag(bb + (size_t)32 * INNER + k0, INNER), acc2);
        acc3 = wmma_f16(a, load_bt_frag(bb + (size_t)48 * INNER + k0, INNER), acc3);
    }
    int ml = (lane >> 4) * 8, nl = lane & 15;
    v8f accs[4] = {acc0, acc1, acc2, acc3};
#pragma unroll
    for (int j = 0; j < 4; ++j) {
        float bv = bias[n0 + j * 16 + nl];
#pragma unroll
        for (int r = 0; r < 8; ++r)
            out[(size_t)(mtile * 16 + ml + r) * DIMM + n0 + j * 16 + nl] = accs[j][r] + bv;
    }
}

// ---------------------------------------------------------------------------
extern "C" void kernel_launch(void* const* d_in, const int* in_sizes, int n_in,
                              void* d_out, int out_size, void* d_ws, size_t ws_size,
                              hipStream_t stream) {
    const float* x        = (const float*)d_in[0];
    const float* ln_gamma = (const float*)d_in[1];
    const float* ln_beta  = (const float*)d_in[2];
    const float* W_qkv    = (const float*)d_in[3];
    const float* W_out    = (const float*)d_in[4];
    const float* b_out    = (const float*)d_in[5];
    float* out = (float*)d_out;

    char* ws = (char*)d_ws;
    size_t off = 0;
    f16* xn     = (f16*)(ws + off); off += (size_t)NTOK * DIMM * 2;      // 8 MB
    f16* wt_qkv = (f16*)(ws + off); off += (size_t)3 * INNER * DIMM * 2; // 6 MB
    f16* wt_out = (f16*)(ws + off); off += (size_t)DIMM * INNER * 2;     // 2 MB
    f16* qbuf   = (f16*)(ws + off); off += (size_t)NTOK * INNER * 2;     // 8 MB
    f16* kbuf   = (f16*)(ws + off); off += (size_t)NTOK * INNER * 2;     // 8 MB
    f16* vtbuf  = (f16*)(ws + off); off += (size_t)NTOK * INNER * 2;     // 8 MB
    f16* obuf   = (f16*)(ws + off); off += (size_t)NTOK * INNER * 2;     // 8 MB

    // 1) LayerNorm -> f16
    ln_to_f16<<<NTOK, 256, 0, stream>>>(x, ln_gamma, ln_beta, xn);

    // 2) weight convert/transpose (f32 [K][N] -> f16 [N][K])
    {
        int tot = DIMM * 3 * INNER;
        wt_transpose<<<(tot + 255) / 256, 256, 0, stream>>>(W_qkv, wt_qkv, DIMM, 3 * INNER);
    }
    {
        int tot = INNER * DIMM;
        wt_transpose<<<(tot + 255) / 256, 256, 0, stream>>>(W_out, wt_out, INNER, DIMM);
    }

    // 3) QKV projection (WMMA, 16x64 per wave)
    qkv_gemm<<<dim3(NTOK / 64, (3 * INNER) / 64), 128, 0, stream>>>(xn, wt_qkv, qbuf, kbuf, vtbuf);

    // 4) flash attention (WMMA + async-to-LDS double buffering)
    attn<<<dim3(SEQ / 64, BSZ * HEADS), 128, 0, stream>>>(qbuf, kbuf, vtbuf, obuf);

    // 5) output projection + bias (WMMA, 16x64 per wave)
    out_gemm<<<dim3(NTOK / 64, DIMM / 64), 128, 0, stream>>>(obuf, wt_out, b_out, out);
}
